// ScanLSTM_40475771797906
// MI455X (gfx1250) — matrix operand order repaired and checked by reading, not compiled, not run on hardware
//
#include <hip/hip_runtime.h>
#include <math.h>

typedef __bf16 bf16_t;
typedef bf16_t v16bf __attribute__((ext_vector_type(16)));
typedef bf16_t v8bf  __attribute__((ext_vector_type(8)));
typedef float  v8f   __attribute__((ext_vector_type(8)));

#define Bx  64
#define Tx  2048
#define Xx  128
#define Hx  256
#define HUx 256
#define Ux  32

// bf16 weight arena offsets (elements) inside d_ws
#define N_WI  (Xx * 4 * Hx)      // 131072
#define N_WH  (Hx * 4 * Hx)      // 262144
#define N_WX1 (Hx * HUx)         // 65536
#define N_WX2 (HUx * Xx)         // 32768
#define N_WU1 (Hx * HUx)         // 65536
#define N_WU2 (HUx * Ux)         // 8192
#define OFF_WI  0
#define OFF_WH  (OFF_WI  + N_WI)
#define OFF_WX1 (OFF_WH  + N_WH)
#define OFF_WX2 (OFF_WX1 + N_WX1)
#define OFF_WU1 (OFF_WX2 + N_WX2)
#define OFF_WU2 (OFF_WU1 + N_WU1)
#define N_WTOT  (OFF_WU2 + N_WU2)   // 565248 bf16 elements (~1.1 MB)

// ---------------------------------------------------------------------------
// Weight prep: fp32 (K x N, row-major) -> bf16 packed in WMMA B-fragment
// order.  Block = (n-tile of 16 cols) x (k-block of 32): 512 contiguous bf16.
//   column n = ntile*16 + (lane & 15)
//   K        = kblock*32 + (lane>>4)*16 + e
// ---------------------------------------------------------------------------
__device__ __forceinline__ void cvt_seg(long long local, const float* src,
                                        int K, int N, bf16_t* dst) {
  int nbk = K >> 5;
  long long blk = local >> 9;
  int e   = (int)(local & 511);
  int nt  = (int)(blk / nbk);
  int kb  = (int)(blk % nbk);
  int l   = e >> 4;
  int j   = e & 15;
  int k   = kb * 32 + ((l >> 4) << 4) + j;
  int n   = nt * 16 + (l & 15);
  dst[local] = (bf16_t)src[(long long)k * N + n];
}

__global__ void prep_weights(const float* Wi, const float* Wh,
                             const float* Wx1, const float* Wx2,
                             const float* Wu1, const float* Wu2, bf16_t* w) {
  long long gid = (long long)blockIdx.x * blockDim.x + threadIdx.x;
  if (gid < OFF_WH)            cvt_seg(gid - OFF_WI,  Wi,  Xx,  4 * Hx, w + OFF_WI);
  else if (gid < OFF_WX1)      cvt_seg(gid - OFF_WH,  Wh,  Hx,  4 * Hx, w + OFF_WH);
  else if (gid < OFF_WX2)      cvt_seg(gid - OFF_WX1, Wx1, Hx,  HUx,    w + OFF_WX1);
  else if (gid < OFF_WU1)      cvt_seg(gid - OFF_WX2, Wx2, HUx, Xx,     w + OFF_WX2);
  else if (gid < OFF_WU2)      cvt_seg(gid - OFF_WU1, Wu1, Hx,  HUx,    w + OFF_WU1);
  else if (gid < N_WTOT)       cvt_seg(gid - OFF_WU2, Wu2, HUx, Ux,     w + OFF_WU2);
}

// ---------------------------------------------------------------------------
// Fragment loaders
// ---------------------------------------------------------------------------
__device__ __forceinline__ v16bf mk16(v8bf lo, v8bf hi) {
  return __builtin_shufflevector(lo, hi, 0, 1, 2, 3, 4, 5, 6, 7,
                                 8, 9, 10, 11, 12, 13, 14, 15);
}

// A fragment (16x32, bf16) from an LDS row-major matrix with row stride S.
__device__ __forceinline__ v16bf loadA(const bf16_t* base, int S, int mt,
                                       int kb, int lane) {
  int row = mt * 16 + (lane & 15);
  int k0  = kb * 32 + ((lane >> 4) << 4);
  const bf16_t* p = base + row * S + k0;
  v8bf lo = *(const v8bf*)(p);
  v8bf hi = *(const v8bf*)(p + 8);
  return mk16(lo, hi);
}

// B fragment (32x16, bf16) from the pre-swizzled global weight arena.
__device__ __forceinline__ v16bf loadB(const bf16_t* wseg, int nbk, int nt,
                                       int kb, int lane) {
  const bf16_t* p = wseg + ((long long)(nt * nbk + kb) << 9) + lane * 16;
  v8bf lo = *(const v8bf*)(p);
  v8bf hi = *(const v8bf*)(p + 8);
  return mk16(lo, hi);
}

__device__ __forceinline__ v8f wmma_bf16(v16bf a, v16bf b, v8f c) {
  return __builtin_amdgcn_wmma_f32_16x16x32_bf16(false, a, false, b,
                                                 (short)0, c, false, false);
}

__device__ __forceinline__ float sigf(float x) {
  return 1.0f / (1.0f + __expf(-x));        // saturates cleanly at +-inf
}
__device__ __forceinline__ float tanh_fast(float x) {
  return 1.0f - 2.0f / (__expf(2.0f * x) + 1.0f);   // nan-free saturation
}

// ---------------------------------------------------------------------------
// Persistent scan kernel: 1 workgroup, 512 threads = 16 wave32 waves.
// All hot loops rolled -> small scheduling scope -> B fragments stay in VGPRs.
// Cell state c lives in LDS; h is double-buffered in LDS.
// ---------------------------------------------------------------------------
#define NTHREADS 512
#define NW 16            // waves per workgroup

__global__ __launch_bounds__(NTHREADS, 1)
void scan_lstm(const float* __restrict__ xseq, const float* __restrict__ c0,
               const float* __restrict__ h0, const int* __restrict__ tfp,
               const float* __restrict__ bh, const float* __restrict__ bx1,
               const float* __restrict__ bx2, const float* __restrict__ bu1,
               const float* __restrict__ bu2, const bf16_t* __restrict__ w,
               float* __restrict__ out) {
  __shared__ bf16_t s_x [Bx * Xx];     // 16 KB  current input, bf16
  __shared__ float  s_xf[Bx * Xx];     // 32 KB  current input / next_x carry
  __shared__ bf16_t s_h0[Bx * Hx];     // 32 KB  hidden state ping
  __shared__ bf16_t s_h1[Bx * Hx];     // 32 KB  hidden state pong
  __shared__ bf16_t s_a1[Bx * HUx];    // 32 KB  relu(h Wx1 + bx1)
  __shared__ bf16_t s_u1[Bx * HUx];    // 32 KB  relu(h Wu1 + bu1)
  __shared__ float  s_c [Bx * Hx];     // 64 KB  cell state, f32
  // total 240 KB of the 320 KB WGP LDS

  const int tid  = threadIdx.x;
  const int wave = tid >> 5;
  const int lane = tid & 31;
  const bool tf = (*tfp) != 0;

  const bf16_t* Wi  = w + OFF_WI;
  const bf16_t* Wh  = w + OFF_WH;
  const bf16_t* Wx1 = w + OFF_WX1;
  const bf16_t* Wx2 = w + OFF_WX2;
  const bf16_t* Wu1 = w + OFF_WU1;
  const bf16_t* Wu2 = w + OFF_WU2;

  float* out_c  = out;
  float* out_h  = out + Bx * Hx;
  float* out_xp = out + 2 * Bx * Hx;
  float* out_xs = out_xp + Bx * Xx;
  float* out_us = out_xs + (long long)Bx * Tx * Xx;

  // ---- init carries ----
  for (int i = tid; i < Bx * Xx; i += NTHREADS) {
    int b = i / Xx, xc = i % Xx;
    s_xf[i] = xseq[(long long)b * Tx * Xx + xc];   // xprev0 = xseq[:,0]
  }
  for (int i = tid; i < Bx * Hx; i += NTHREADS) {
    s_h0[i] = (bf16_t)h0[i];
    s_c[i]  = c0[i];
  }
  __syncthreads();

  for (int t = 0; t < Tx; ++t) {
    const bf16_t* hcur = (t & 1) ? s_h1 : s_h0;
    bf16_t*       hnxt = (t & 1) ? s_h0 : s_h1;

    // -------- phase 1: select input x_t, publish bf16 copy --------
    #pragma unroll 1
    for (int i = tid; i < Bx * Xx; i += NTHREADS) {
      float xv;
      if (tf) {
        int b = i / Xx, xc = i % Xx;
        xv = xseq[((long long)b * Tx + t) * Xx + xc];
        s_xf[i] = xv;
      } else {
        xv = s_xf[i];                  // next_x from previous step
      }
      s_x[i] = (bf16_t)xv;
    }
    __syncthreads();

    // -------- phase 2: gates = x@Wi + h@Wh + bh, LSTM cell --------
    // writes h_new into the OTHER h buffer; c read+written in LDS
    #pragma unroll 1
    for (int s = 0; s < 4; ++s) {
      int job = wave + s * NW;
      int mt = job >> 4, nh = job & 15;
      int col = nh * 16 + (lane & 15);
      v8f acc[4];
      #pragma unroll
      for (int g = 0; g < 4; ++g) {
        float bv = bh[g * Hx + col];
        #pragma unroll
        for (int r = 0; r < 8; ++r) acc[g][r] = bv;
      }
      #pragma unroll 1
      for (int kb = 0; kb < 4; ++kb) {              // K = 128 from x
        v16bf a = loadA(s_x, Xx, mt, kb, lane);
        #pragma unroll
        for (int g = 0; g < 4; ++g)
          acc[g] = wmma_bf16(a, loadB(Wi, 4, g * 16 + nh, kb, lane), acc[g]);
      }
      #pragma unroll 1
      for (int kb = 0; kb < 8; ++kb) {              // K = 256 from h
        v16bf a = loadA(hcur, Hx, mt, kb, lane);
        #pragma unroll
        for (int g = 0; g < 4; ++g)
          acc[g] = wmma_bf16(a, loadB(Wh, 8, g * 16 + nh, kb, lane), acc[g]);
      }
      #pragma unroll
      for (int r = 0; r < 8; ++r) {                 // cell update (i,f,g,o)
        int m = mt * 16 + r + ((lane >> 4) << 3);
        float iv = sigf(acc[0][r]);
        float fv = sigf(acc[1][r]);
        float gv = tanh_fast(acc[2][r]);
        float ov = sigf(acc[3][r]);
        float cn = fv * s_c[m * Hx + col] + iv * gv;
        float hv = ov * tanh_fast(cn);
        s_c[m * Hx + col] = cn;
        hnxt[m * Hx + col] = (bf16_t)hv;
        if (t == Tx - 1) out_h[m * Hx + col] = hv;
      }
    }
    __syncthreads();   // h_new visible to all waves

    // -------- phase 3: a1 = relu(h' Wx1+bx1), u1 = relu(h' Wu1+bu1) --------
    #pragma unroll 1
    for (int job = wave; job < 128; job += NW) {
      int sel = job >> 6;
      int sub = job & 63;
      int mt = sub >> 4, nt = sub & 15;
      int col = nt * 16 + (lane & 15);
      const bf16_t* Wp = sel ? Wu1 : Wx1;
      const float*  bp = sel ? bu1 : bx1;
      bf16_t* dst = sel ? s_u1 : s_a1;
      v8f acc;
      {
        float bv = bp[col];
        #pragma unroll
        for (int r = 0; r < 8; ++r) acc[r] = bv;
      }
      #pragma unroll
      for (int kb = 0; kb < 8; ++kb) {
        v16bf a = loadA(hnxt, Hx, mt, kb, lane);
        acc = wmma_bf16(a, loadB(Wp, 8, nt, kb, lane), acc);
      }
      #pragma unroll
      for (int r = 0; r < 8; ++r) {
        int m = mt * 16 + r + ((lane >> 4) << 3);
        dst[m * HUx + col] = (bf16_t)fmaxf(acc[r], 0.0f);
      }
    }
    __syncthreads();

    // -------- phase 4: next_x = a1 Wx2 + bx2 + x ; u = tanh(u1 Wu2 + bu2)
    #pragma unroll 1
    for (int job = wave; job < 40; job += NW) {
      if (job < 32) {
        int mt = job >> 3, nt = job & 7;
        int col = nt * 16 + (lane & 15);
        v8f acc;
        {
          float bv = bx2[col];
          #pragma unroll
          for (int r = 0; r < 8; ++r) acc[r] = bv;
        }
        #pragma unroll
        for (int kb = 0; kb < 8; ++kb) {
          v16bf a = loadA(s_a1, HUx, mt, kb, lane);
          acc = wmma_bf16(a, loadB(Wx2, 8, nt, kb, lane), acc);
        }
        #pragma unroll
        for (int r = 0; r < 8; ++r) {
          int m = mt * 16 + r + ((lane >> 4) << 3);
          float v = acc[r] + s_xf[m * Xx + col];    // residual with f32 x
          s_xf[m * Xx + col] = v;                   // becomes next step's carry
          out_xs[((long long)m * Tx + t) * Xx + col] = v;
        }
      } else {
        int j = job - 32;
        int mt = j >> 1, nt = j & 1;
        int col = nt * 16 + (lane & 15);
        v8f acc;
        {
          float bv = bu2[col];
          #pragma unroll
          for (int r = 0; r < 8; ++r) acc[r] = bv;
        }
        #pragma unroll
        for (int kb = 0; kb < 8; ++kb) {
          v16bf a = loadA(s_u1, HUx, mt, kb, lane);
          acc = wmma_bf16(a, loadB(Wu2, 8, nt, kb, lane), acc);
        }
        #pragma unroll
        for (int r = 0; r < 8; ++r) {
          int m = mt * 16 + r + ((lane >> 4) << 3);
          out_us[((long long)m * Tx + t) * Ux + col] = tanh_fast(acc[r]);
        }
      }
    }
    __syncthreads();   // protect s_xf / s_x reuse next iteration
  }

  // final carries: xprev = last next_x, c from LDS
  for (int i = tid; i < Bx * Xx; i += NTHREADS) out_xp[i] = s_xf[i];
  for (int i = tid; i < Bx * Hx; i += NTHREADS) out_c[i] = s_c[i];
}

extern "C" void kernel_launch(void* const* d_in, const int* in_sizes, int n_in,
                              void* d_out, int out_size, void* d_ws,
                              size_t ws_size, hipStream_t stream) {
  const float* xseq = (const float*)d_in[0];
  const float* c0   = (const float*)d_in[1];
  const float* h0   = (const float*)d_in[2];
  const int*   tf   = (const int*)d_in[3];
  const float* Wi   = (const float*)d_in[4];
  const float* Wh   = (const float*)d_in[5];
  const float* bh   = (const float*)d_in[6];
  const float* Wx1  = (const float*)d_in[7];
  const float* bx1  = (const float*)d_in[8];
  const float* Wx2  = (const float*)d_in[9];
  const float* bx2  = (const float*)d_in[10];
  const float* Wu1  = (const float*)d_in[11];
  const float* bu1  = (const float*)d_in[12];
  const float* Wu2  = (const float*)d_in[13];
  const float* bu2  = (const float*)d_in[14];
  bf16_t* w = (bf16_t*)d_ws;

  // one-shot bf16 pack+swizzle of all weights (stays hot in the 192MB L2)
  prep_weights<<<(N_WTOT + 255) / 256, 256, 0, stream>>>(Wi, Wh, Wx1, Wx2,
                                                         Wu1, Wu2, w);
  // persistent single-workgroup sequential scan
  scan_lstm<<<1, NTHREADS, 0, stream>>>(xseq, c0, h0, tf, bh, bx1, bx2, bu1,
                                        bu2, w, (float*)d_out);
}
